// WeiMultiHeadSelfAttention_28535762715187
// MI455X (gfx1250) — compile-verified, matrix operand order checked by reading, not run on hardware
//
#include <hip/hip_runtime.h>
#include <hip/hip_bf16.h>
#include <stdint.h>

// ---------------------------------------------------------------------------
// MI455X (gfx1250) multi-head self-attention (heads NOT split, per reference).
// GEMMs: v_wmma_f32_16x16x32_bf16, LDS-tiled, double-buffered async copies
// (global_load_async_to_lds_b128 + s_wait_asynccnt).
// ---------------------------------------------------------------------------

#define BATCH 4
#define SEQ   2048
#define DIM   1024   // == NUM_HEADS * D_Q == d_model

#define BM 128       // block tile M
#define BN 128       // block tile N
#define BK 64        // block K step
#define LPITCH 144   // LDS row pitch in bytes (128B data + 16B pad, 16B aligned)

typedef __attribute__((ext_vector_type(16))) __bf16 v16bf;
typedef __attribute__((ext_vector_type(8)))  float  v8f;

union V16 { v16bf v; uint4 q[2]; };

static __device__ __forceinline__ unsigned short f2bf_bits(float f) {
    // round-to-nearest-even f32 -> bf16 (bit pattern)
    unsigned u = __float_as_uint(f);
    u = (u + 0x7FFFu + ((u >> 16) & 1u)) >> 16;
    return (unsigned short)u;
}

// One per-lane 16-byte async copy: global -> LDS (tracked by ASYNCcnt).
static __device__ __forceinline__ void async_cp_b128(const void* g, void* l) {
    unsigned lofs = (unsigned)(unsigned long long)
        ((__attribute__((address_space(3))) unsigned char*)l);
    unsigned long long ga = (unsigned long long)(uintptr_t)g;
    asm volatile("global_load_async_to_lds_b128 %0, %1, off"
                 :: "v"(lofs), "v"(ga) : "memory");
}

// --------------------------- f32 -> bf16 convert ---------------------------
__global__ __launch_bounds__(256) void cvt_f32_to_bf16(
    const float* __restrict__ in, unsigned short* __restrict__ out, long n)
{
    long i = ((long)blockIdx.x * 256 + threadIdx.x) * 4;
    if (i >= n) return;
    float4 f = *(const float4*)(in + i);
    uint2 pk;
    pk.x = (unsigned)f2bf_bits(f.x) | ((unsigned)f2bf_bits(f.y) << 16);
    pk.y = (unsigned)f2bf_bits(f.z) | ((unsigned)f2bf_bits(f.w) << 16);
    *(uint2*)(out + i) = pk;
}

// ------------------------------- NT GEMM -----------------------------------
// C[M,N] = scale * A[M,K](bf16,row-major,pitch lda) * B[N,K]^T(bf16,row-major,
// pitch ldb).  Block 128x128, 8 waves in 4x2, wave tile 32x64 (8 WMMA accs).
// A/B staged in LDS via double-buffered async copies.
enum { OUT_F32 = 0, OUT_BF16 = 1, OUT_BF16T = 2 };

// 256 threads stage one 128 x BK strip (8 x 16B chunks per row) -> 4 asyncs/thread
static __device__ __forceinline__ void issue_strip(
    const __bf16* gBase, long ld, int kStep, unsigned char* lbuf, int tid)
{
    const int r  = tid >> 1;            // 0..127 row in strip
    const int c0 = (tid & 1) * 4;       // chunk 0 or 4
    const __bf16*  g = gBase + (long)r * ld + kStep + c0 * 8;
    unsigned char* l = lbuf + r * LPITCH + c0 * 16;
#pragma unroll
    for (int j = 0; j < 4; ++j)
        async_cp_b128(g + j * 8, l + j * 16);
}

template<int OUT_MODE, bool CSKIP, bool CKLIM>
__global__ __launch_bounds__(256) void gemm_nt(
    const __bf16* __restrict__ A, long lda, long aBatch,
    const __bf16* __restrict__ B, long ldb, long bBatch,
    void* __restrict__ Cv, long ldc, long cBatch,
    int K, float scale, long tRows)
{
    __shared__ __align__(16) unsigned char sA[2][BM * LPITCH];
    __shared__ __align__(16) unsigned char sB[2][BN * LPITCH];

    const int  tid = threadIdx.x;
    const int  w   = tid >> 5;
    const int  wm  = w >> 1;                 // 0..3  (32-row slice)
    const int  wn  = w & 1;                  // 0..1  (64-col slice)
    const int  lh  = tid & 15;               // lane % 16
    const int  hi  = (tid >> 4) & 1;         // lane / 16
    const long m0  = (long)blockIdx.y * BM;
    const long n0  = (long)blockIdx.x * BN;
    const long z   = blockIdx.z;

    if (CSKIP && n0 > m0 + (BM - 1)) return;   // block fully above causal diag

    int kmax = K;
    if (CKLIM) {
        long kl = m0 + BM;                     // multiple of BK already
        kmax = (int)(kl < (long)K ? kl : (long)K);
    }

    const __bf16* gA = A + z * aBatch + m0 * lda;
    const __bf16* gB = B + z * bBatch + n0 * ldb;

    v8f acc[2][4] = {};

    // prologue: fill buffer 0
    issue_strip(gA, lda, 0, sA[0], tid);
    issue_strip(gB, ldb, 0, sB[0], tid);

    int buf = 0;
    for (int k0 = 0; k0 < kmax; k0 += BK, buf ^= 1) {
        const bool more = (k0 + BK) < kmax;
        if (more) {
            issue_strip(gA, lda, k0 + BK, sA[buf ^ 1], tid);
            issue_strip(gB, ldb, k0 + BK, sB[buf ^ 1], tid);
            asm volatile("s_wait_asynccnt 0x8" ::: "memory");
        } else {
            asm volatile("s_wait_asynccnt 0x0" ::: "memory");
        }
        __syncthreads();                       // current buffer visible to all

        const unsigned char* la = sA[buf];
        const unsigned char* lb = sB[buf];
#pragma unroll
        for (int kk = 0; kk < BK; kk += 32) {
            V16 af[2];
#pragma unroll
            for (int tm = 0; tm < 2; ++tm) {
                const int row = wm * 32 + tm * 16 + lh;
                const unsigned char* ap = la + row * LPITCH + kk * 2 + hi * 16;
                af[tm].q[0] = *(const uint4*)ap;
                af[tm].q[1] = *(const uint4*)(ap + 32);
            }
#pragma unroll
            for (int tn = 0; tn < 4; ++tn) {
                const int col = wn * 64 + tn * 16 + lh;
                const unsigned char* bp = lb + col * LPITCH + kk * 2 + hi * 16;
                V16 bf_;
                bf_.q[0] = *(const uint4*)bp;
                bf_.q[1] = *(const uint4*)(bp + 32);
                acc[0][tn] = __builtin_amdgcn_wmma_f32_16x16x32_bf16(
                    false, af[0].v, false, bf_.v, (short)0, acc[0][tn], false, false);
                acc[1][tn] = __builtin_amdgcn_wmma_f32_16x16x32_bf16(
                    false, af[1].v, false, bf_.v, (short)0, acc[1][tn], false, false);
            }
        }
        __syncthreads();                       // reads done before next overwrite
    }

    // C/D f32 layout: vgpr i holds (mBase + i, n)
    const long mW = m0 + wm * 32;
    const long nW = n0 + wn * 64;
    if (OUT_MODE == OUT_F32) {
        float* C = (float*)Cv + z * cBatch;
#pragma unroll
        for (int tm = 0; tm < 2; ++tm) {
            const long mBase = mW + tm * 16 + hi * 8;
#pragma unroll
            for (int tn = 0; tn < 4; ++tn) {
                const long n = nW + tn * 16 + lh;
#pragma unroll
                for (int i = 0; i < 8; ++i)
                    C[(mBase + i) * ldc + n] = acc[tm][tn][i] * scale;
            }
        }
    } else if (OUT_MODE == OUT_BF16) {
        unsigned short* C = (unsigned short*)Cv + z * cBatch;
#pragma unroll
        for (int tm = 0; tm < 2; ++tm) {
            const long mBase = mW + tm * 16 + hi * 8;
#pragma unroll
            for (int tn = 0; tn < 4; ++tn) {
                const long n = nW + tn * 16 + lh;
#pragma unroll
                for (int i = 0; i < 8; ++i)
                    C[(mBase + i) * ldc + n] = f2bf_bits(acc[tm][tn][i] * scale);
            }
        }
    } else { // OUT_BF16T: transposed store; 8 M-rows -> one contiguous b128
#pragma unroll
        for (int tm = 0; tm < 2; ++tm) {
            const long mBase = mW + tm * 16 + hi * 8;
            const long bt = mBase / tRows;
            const long ml = mBase - bt * tRows;
            unsigned short* C = (unsigned short*)Cv + bt * cBatch;
#pragma unroll
            for (int tn = 0; tn < 4; ++tn) {
                const long n = nW + tn * 16 + lh;
                unsigned short h[8];
#pragma unroll
                for (int i = 0; i < 8; ++i) h[i] = f2bf_bits(acc[tm][tn][i] * scale);
                uint4 pk;
                pk.x = (unsigned)h[0] | ((unsigned)h[1] << 16);
                pk.y = (unsigned)h[2] | ((unsigned)h[3] << 16);
                pk.z = (unsigned)h[4] | ((unsigned)h[5] << 16);
                pk.w = (unsigned)h[6] | ((unsigned)h[7] << 16);
                *(uint4*)(C + n * ldc + ml) = pk;
            }
        }
    }
}

// --------------------- causal softmax, bf16 out in-place -------------------
__global__ __launch_bounds__(256) void softmax_causal_inplace(float* __restrict__ Sf)
{
    __shared__ float red[256];
    const int  tid = threadIdx.x;
    const long row = blockIdx.x;               // 0 .. BATCH*SEQ-1
    const int  q   = (int)(row & (SEQ - 1));
    float* srow = Sf + row * (long)SEQ;
    const int L = q + 1;

    float v[SEQ / 256];
    float m = -__builtin_inff();
#pragma unroll
    for (int t = 0; t < SEQ / 256; ++t) {
        int j = tid + t * 256;
        float x = (j < L) ? srow[j] : -__builtin_inff();
        v[t] = x;
        m = fmaxf(m, x);
    }
    red[tid] = m; __syncthreads();
    for (int s2 = 128; s2 > 0; s2 >>= 1) {
        if (tid < s2) red[tid] = fmaxf(red[tid], red[tid + s2]);
        __syncthreads();
    }
    const float M = red[0];
    __syncthreads();

    float sum = 0.f;
#pragma unroll
    for (int t = 0; t < SEQ / 256; ++t) {
        int j = tid + t * 256;
        float e = (j < L) ? __expf(v[t] - M) : 0.f;
        v[t] = e;
        sum += e;
    }
    red[tid] = sum; __syncthreads();
    for (int s2 = 128; s2 > 0; s2 >>= 1) {
        if (tid < s2) red[tid] += red[tid + s2];
        __syncthreads();
    }
    const float rinv = 1.0f / red[0];
    __syncthreads();    // all f32 reads complete before bf16 overwrite

    unsigned short* arow = (unsigned short*)srow;  // att row, pitch 2*SEQ bf16
#pragma unroll
    for (int t = 0; t < SEQ / 256; ++t) {
        int j = tid + t * 256;
        arow[j] = f2bf_bits(v[t] * rinv);          // 0 for masked positions
    }
}

// ------------------------------- launcher ----------------------------------
extern "C" void kernel_launch(void* const* d_in, const int* in_sizes, int n_in,
                              void* d_out, int out_size, void* d_ws, size_t ws_size,
                              hipStream_t stream)
{
    (void)in_sizes; (void)n_in; (void)out_size; (void)ws_size;

    const float* X  = (const float*)d_in[0];
    const float* Wq = (const float*)d_in[1];
    const float* Wk = (const float*)d_in[2];
    const float* Wv = (const float*)d_in[3];
    const float* Wo = (const float*)d_in[4];

    char* p = (char*)d_ws;
    auto take = [&](size_t bytes) {
        char* r = p; p += (bytes + 255) & ~(size_t)255; return r;
    };

    const size_t nX = (size_t)BATCH * SEQ * DIM;   // 8,388,608
    const size_t nW = (size_t)DIM * DIM;           // 1,048,576

    unsigned short* Xb  = (unsigned short*)take(nX * 2);   // X bf16 (reused for ctx)
    unsigned short* Wqb = (unsigned short*)take(nW * 2);
    unsigned short* Wkb = (unsigned short*)take(nW * 2);
    unsigned short* Wvb = (unsigned short*)take(nW * 2);
    unsigned short* Wob = (unsigned short*)take(nW * 2);
    unsigned short* Qb  = (unsigned short*)take(nX * 2);
    unsigned short* Kb  = (unsigned short*)take(nX * 2);
    unsigned short* Vtb = (unsigned short*)take(nX * 2);   // V^T per batch [DIM, SEQ]
    float*          Sf  = (float*)take((size_t)BATCH * SEQ * SEQ * 4);
    unsigned short* Cb  = Xb;                              // ctx bf16, reuse X scratch

    const dim3 blk(256);

    // 1) convert to bf16
    cvt_f32_to_bf16<<<dim3((unsigned)(nX / (256 * 4))), blk, 0, stream>>>(X,  Xb,  (long)nX);
    cvt_f32_to_bf16<<<dim3((unsigned)(nW / (256 * 4))), blk, 0, stream>>>(Wq, Wqb, (long)nW);
    cvt_f32_to_bf16<<<dim3((unsigned)(nW / (256 * 4))), blk, 0, stream>>>(Wk, Wkb, (long)nW);
    cvt_f32_to_bf16<<<dim3((unsigned)(nW / (256 * 4))), blk, 0, stream>>>(Wv, Wvb, (long)nW);
    cvt_f32_to_bf16<<<dim3((unsigned)(nW / (256 * 4))), blk, 0, stream>>>(Wo, Wob, (long)nW);

    // 2) projections (M = BATCH*SEQ folded, weights shared)
    const dim3 gProj(DIM / BN, (BATCH * SEQ) / BM, 1);
    gemm_nt<OUT_BF16, false, false><<<gProj, blk, 0, stream>>>(
        (const __bf16*)Xb, DIM, 0, (const __bf16*)Wqb, DIM, 0,
        Qb, DIM, 0, DIM, 1.0f, 1);
    gemm_nt<OUT_BF16, false, false><<<gProj, blk, 0, stream>>>(
        (const __bf16*)Xb, DIM, 0, (const __bf16*)Wkb, DIM, 0,
        Kb, DIM, 0, DIM, 1.0f, 1);
    gemm_nt<OUT_BF16T, false, false><<<gProj, blk, 0, stream>>>(
        (const __bf16*)Xb, DIM, 0, (const __bf16*)Wvb, DIM, 0,
        Vtb, SEQ /*ldT*/, (long)DIM * SEQ /*batch stride*/, DIM, 1.0f, SEQ);

    // 3) scores = Q K^T / 32 (f32), skipping blocks above the causal diagonal
    gemm_nt<OUT_F32, true, false><<<dim3(SEQ / BN, SEQ / BM, BATCH), blk, 0, stream>>>(
        (const __bf16*)Qb, DIM, (long)SEQ * DIM,
        (const __bf16*)Kb, DIM, (long)SEQ * DIM,
        Sf, SEQ, (long)SEQ * SEQ, DIM, 0.03125f, 1);

    // 4) causal softmax, bf16 att written in-place (pitch 2*SEQ bf16)
    softmax_causal_inplace<<<dim3(BATCH * SEQ), blk, 0, stream>>>(Sf);

    // 5) ctx = att @ V  (B = V^T [DIM, SEQ]); causal K-limit per block
    gemm_nt<OUT_BF16, false, true><<<dim3(DIM / BN, SEQ / BM, BATCH), blk, 0, stream>>>(
        (const __bf16*)Sf, 2 * SEQ, (long)SEQ * 2 * SEQ,
        (const __bf16*)Vtb, SEQ, (long)DIM * SEQ,
        Cb, DIM, (long)SEQ * DIM, SEQ, 1.0f, 1);

    // 6) out = ctx @ Wo^T (f32 -> d_out)
    gemm_nt<OUT_F32, false, false><<<gProj, blk, 0, stream>>>(
        (const __bf16*)Cb, DIM, 0, (const __bf16*)Wob, DIM, 0,
        (float*)d_out, DIM, 0, DIM, 1.0f, 1);
}
